// MultiHeadAttention_17128329576953
// MI455X (gfx1250) — matrix operand order repaired and checked by reading, not compiled, not run on hardware
//
#include <hip/hip_runtime.h>
#include <hip/hip_bf16.h>

// MI455X / gfx1250, wave32. All matrix math via v_wmma_f32_16x16x32_f16
// (f16 A/B operands, f32 accumulation). Software-pipelined; where tiles are
// pure f16 copies (attention k/v, out-proj A) we use the CDNA5 async
// global->LDS path (GLOBAL_LOAD_ASYNC_TO_LDS_B128 + s_wait_asynccnt) with
// double-buffered LDS. Falls back to VGPR-staged copies if the async
// builtins are unavailable.

typedef __attribute__((ext_vector_type(16))) _Float16 v16h;
typedef __attribute__((ext_vector_type(8)))  _Float16 v8h;
typedef __attribute__((ext_vector_type(4)))  _Float16 v4h;
typedef __attribute__((ext_vector_type(8)))  float    v8f;
typedef __attribute__((ext_vector_type(4)))  int      v4i;

// v4i living in global (AS1) / LDS (AS3) address spaces, as required by the
// async-copy builtin's prototype: (v4i AS1*, v4i AS3*, imm offset, imm cpol).
typedef v4i __attribute__((address_space(1))) gv4i;
typedef v4i __attribute__((address_space(3))) lv4i;

#if defined(__HIP_DEVICE_COMPILE__) && defined(__has_builtin)
#if __has_builtin(__builtin_amdgcn_global_load_async_to_lds_b128) && \
    __has_builtin(__builtin_amdgcn_s_wait_asynccnt)
#define USE_ASYNC_LDS 1
#endif
#endif
#ifndef USE_ASYNC_LDS
#define USE_ASYNC_LDS 0
#endif

#if USE_ASYNC_LDS
#define ASYNC_CP_B128(gsrc, ldst)                                              \
  __builtin_amdgcn_global_load_async_to_lds_b128(                              \
      (gv4i*)(gsrc), (lv4i*)(ldst), 0, 0)
#define KVBUF 2
#else
#define KVBUF 1
#endif

#define WMMA_F16(a, b, c) \
  __builtin_amdgcn_wmma_f32_16x16x32_f16(false, (a), false, (b), (short)0, (c), false, false)

static __device__ __forceinline__ v16h cat16(v8h lo, v8h hi) {
  return __builtin_shufflevector(lo, hi, 0,1,2,3,4,5,6,7,8,9,10,11,12,13,14,15);
}

#define TSEQ  2048
#define NH    16
#define CDIM  1024
#define DHEAD 64
#define NEGBIG (-1.0e30f)
// 0.125 * log2(e): fold softmax scale + exp->exp2 conversion into q.
#define QSCALE 0.18033688011112042f

// LDS row strides (in halves) chosen for conflict-free ds_load_b128 fragment reads.
#define LDA32 40   // rows of 32 halves padded to 40 (80B -> 20-bank stride, full 64-bank coverage)
#define LDA64 72   // rows of 64 halves padded to 72 (144B -> 36-bank stride, full coverage)

// ---------------------------------------------------------------------------
// Kernel 1: per-head QKV projections.  grid.x = 3 * NH * 32
//   which = 0: q = Q @ Wq[h]   (scaled by QSCALE, stored [B][H][T][64] f16)
//   which = 1: k = K @ Wk[h]   (stored [B][H][T][64] f16)
//   which = 2: v = V @ Wv[h]   (stored TRANSPOSED [B][H][64][T] f16)
// f32->f16 conversion required, so tiles stay VGPR-staged (pipelined).
// ---------------------------------------------------------------------------
__global__ __launch_bounds__(256) void proj_qkv(
    const float* __restrict__ Kin, const float* __restrict__ Qin, const float* __restrict__ Vin,
    const float* __restrict__ Wk,  const float* __restrict__ Wq,  const float* __restrict__ Wv,
    _Float16* __restrict__ qws, _Float16* __restrict__ kws, _Float16* __restrict__ vtws)
{
  __shared__ _Float16 Ash[128 * LDA32];
  __shared__ _Float16 Bsh[64 * LDA32];

  const int bid   = blockIdx.x;
  const int mb    = bid & 31;
  const int h     = (bid >> 5) & 15;
  const int which = bid >> 9;              // 0=q, 1=k, 2=v

  const float* X = (which == 0) ? Qin : (which == 1) ? Kin : Vin;
  const float* W = ((which == 0) ? Wq : (which == 1) ? Wk : Wv) + (size_t)h * CDIM * DHEAD;

  const int m0   = mb * 128;
  const int tid  = threadIdx.x;
  const int lane = tid & 31;
  const int wv   = __builtin_amdgcn_readfirstlane(tid >> 5);
  const int l16  = lane & 15;
  const int hs   = lane >> 4;

  v8f acc[4] = {};
  float4 aPf[4], bPf[2];

  auto load_tiles = [&](int k0) {
    #pragma unroll
    for (int j = 0; j < 4; ++j) {
      const int idx = tid + j * 256, r = idx >> 3, seg = idx & 7;
      aPf[j] = *(const float4*)(X + (size_t)(m0 + r) * CDIM + k0 + seg * 4);
    }
    #pragma unroll
    for (int j = 0; j < 2; ++j) {
      const int idx = tid + j * 256, kr = idx >> 4, ns = idx & 15;
      bPf[j] = *(const float4*)(W + (size_t)(k0 + kr) * DHEAD + ns * 4);
    }
  };
  auto store_tiles = [&]() {
    #pragma unroll
    for (int j = 0; j < 4; ++j) {
      const int idx = tid + j * 256, r = idx >> 3, seg = idx & 7;
      v4h h4;
      h4[0] = (_Float16)aPf[j].x; h4[1] = (_Float16)aPf[j].y;
      h4[2] = (_Float16)aPf[j].z; h4[3] = (_Float16)aPf[j].w;
      *(v4h*)&Ash[r * LDA32 + seg * 4] = h4;
    }
    #pragma unroll
    for (int j = 0; j < 2; ++j) {
      const int idx = tid + j * 256, kr = idx >> 4, ns = idx & 15;
      Bsh[(ns * 4 + 0) * LDA32 + kr] = (_Float16)bPf[j].x;
      Bsh[(ns * 4 + 1) * LDA32 + kr] = (_Float16)bPf[j].y;
      Bsh[(ns * 4 + 2) * LDA32 + kr] = (_Float16)bPf[j].z;
      Bsh[(ns * 4 + 3) * LDA32 + kr] = (_Float16)bPf[j].w;
    }
  };

  load_tiles(0);
  for (int k0 = 0; k0 < CDIM; k0 += 32) {
    store_tiles();
    __syncthreads();
    if (k0 + 32 < CDIM) load_tiles(k0 + 32);   // overlap with WMMAs below

    const int ar = wv * 16 + l16;
    const int ab = hs * 8;
    v16h afrag = cat16(*(const v8h*)&Ash[ar * LDA32 + ab],
                       *(const v8h*)&Ash[ar * LDA32 + ab + 16]);
    v16h bfr[4];
    #pragma unroll
    for (int nt = 0; nt < 4; ++nt) {
      const int bn = nt * 16 + l16;
      const int bb = hs * 8;
      bfr[nt] = cat16(*(const v8h*)&Bsh[bn * LDA32 + bb],
                      *(const v8h*)&Bsh[bn * LDA32 + bb + 16]);
    }
    #pragma unroll
    for (int nt = 0; nt < 4; ++nt)
      acc[nt] = WMMA_F16(afrag, bfr[nt], acc[nt]);
    __syncthreads();
  }

  // Epilogue
  if (which < 2) {
    const float scale = (which == 0) ? QSCALE : 1.0f;
    _Float16* dst = (which == 0) ? qws : kws;
    const int mloc = wv * 16 + hs * 8;
    #pragma unroll
    for (int nt = 0; nt < 4; ++nt) {
      const int c = nt * 16 + l16;
      #pragma unroll
      for (int v2 = 0; v2 < 8; ++v2) {
        const int mf = m0 + mloc + v2;
        const int bb2 = mf >> 11, t = mf & (TSEQ - 1);
        dst[((size_t)(bb2 * NH + h) * TSEQ + t) * DHEAD + c] = (_Float16)(acc[nt][v2] * scale);
      }
    }
  } else {
    // v transposed: vt[b][h][d][t]; lane holds 8 consecutive t's -> one b128 store per tile
    const int mf  = m0 + wv * 16 + hs * 8;
    const int bb2 = mf >> 11, t0 = mf & (TSEQ - 1);
    #pragma unroll
    for (int nt = 0; nt < 4; ++nt) {
      const int d = nt * 16 + l16;
      v8h pk;
      #pragma unroll
      for (int v2 = 0; v2 < 8; ++v2) pk[v2] = (_Float16)acc[nt][v2];
      *(v8h*)&vtws[((size_t)(bb2 * NH + h) * DHEAD + d) * TSEQ + t0] = pk;
    }
  }
}

// ---------------------------------------------------------------------------
// Kernel 2: causal flash attention.  grid.x = B * NH * (T/128)
// One workgroup per (b, h, 128-row q block); wave owns 16 q rows.
// Computes S^T = k @ q^T so softmax stats are per-lane scalars, q stays in regs.
// k/v^T tiles are pure f16 copies -> async global->LDS with double buffering
// (4 async b128 per wave per tile; in-order completion => s_wait_asynccnt 4
// guarantees the current tile while the next is still in flight).
// ---------------------------------------------------------------------------
__global__ __launch_bounds__(256) void attn(
    const _Float16* __restrict__ qws, const _Float16* __restrict__ kws,
    const _Float16* __restrict__ vtws, _Float16* __restrict__ aws)
{
  __shared__ _Float16 ksh[KVBUF][64 * LDA64];   // k tile   [s][d]
  __shared__ _Float16 vsh[KVBUF][64 * LDA64];   // v^T tile [d][s]

  const int bid = blockIdx.x;
  const int qb  = bid & 15;
  const int h   = (bid >> 4) & 15;
  const int b   = bid >> 8;
  const int bh  = b * NH + h;
  const int q0  = qb * 128;

  const int tid  = threadIdx.x;
  const int lane = tid & 31;
  const int wv   = __builtin_amdgcn_readfirstlane(tid >> 5);
  const int l16  = lane & 15;
  const int hs   = lane >> 4;

  const _Float16* qbase = qws  + (size_t)bh * TSEQ * DHEAD;
  const _Float16* kbase = kws  + (size_t)bh * TSEQ * DHEAD;
  const _Float16* vbase = vtws + (size_t)bh * DHEAD * TSEQ;

  const int tw = q0 + wv * 16;           // wave's first q row (scalar)
  const int tq = tw + l16;               // this lane's q column (B-frag N)

  // q B-fragments, persistent: lanes<16 K=ks*32+0..15, lanes>=16 K=ks*32+16..31
  v16h qf[2];
  #pragma unroll
  for (int ks = 0; ks < 2; ++ks) {
    const v8h* p = (const v8h*)(qbase + (size_t)tq * DHEAD + ks * 32 + hs * 16);
    qf[ks] = cat16(p[0], p[1]);
  }

  v8f O[4] = {};                         // out^T frags: (d-tile) x [d rows, t cols]
  float mrow = NEGBIG, lrow = 0.0f;

#if USE_ASYNC_LDS
  auto issue_kv = [&](int s0, int buf) {
    #pragma unroll
    for (int j = 0; j < 2; ++j) {
      const int idx = tid + j * 256, r = idx >> 3, sg = idx & 7;
      ASYNC_CP_B128(kbase + (size_t)(s0 + r) * DHEAD + sg * 8,
                    &ksh[buf][r * LDA64 + sg * 8]);
      ASYNC_CP_B128(vbase + (size_t)r * TSEQ + s0 + sg * 8,
                    &vsh[buf][r * LDA64 + sg * 8]);
    }
  };
#else
  v8h kPf[2], vPf[2];
  auto load_kv = [&](int s0) {
    #pragma unroll
    for (int j = 0; j < 2; ++j) {
      const int idx = tid + j * 256, r = idx >> 3, sg = idx & 7;
      kPf[j] = *(const v8h*)(kbase + (size_t)(s0 + r) * DHEAD + sg * 8);
      vPf[j] = *(const v8h*)(vbase + (size_t)r * TSEQ + s0 + sg * 8);
    }
  };
  auto store_kv = [&]() {
    #pragma unroll
    for (int j = 0; j < 2; ++j) {
      const int idx = tid + j * 256, r = idx >> 3, sg = idx & 7;
      *(v8h*)&ksh[0][r * LDA64 + sg * 8] = kPf[j];
      *(v8h*)&vsh[0][r * LDA64 + sg * 8] = vPf[j];
    }
  };
#endif

  const int nkb = (q0 >> 6) + 2;         // key blocks of 64 covering [0, q0+128)
#if USE_ASYNC_LDS
  issue_kv(0, 0);
#else
  load_kv(0);
#endif
  for (int kb = 0; kb < nkb; ++kb) {
    const int s0 = kb * 64;
#if USE_ASYNC_LDS
    const int cur = kb & 1;
    if (kb + 1 < nkb) {
      issue_kv(s0 + 64, cur ^ 1);                 // prior reads of that buffer
      __builtin_amdgcn_s_wait_asynccnt(4);        // finished at last barrier
    } else {
      __builtin_amdgcn_s_wait_asynccnt(0);
    }
    __syncthreads();
#else
    const int cur = 0;
    store_kv();
    __syncthreads();
    if (kb + 1 < nkb) load_kv(s0 + 64);  // overlap next tile with this block's math
#endif
    const _Float16* kcur = ksh[cur];
    const _Float16* vcur = vsh[cur];

    if (s0 <= tw + 15) {                 // wave-uniform: skip fully-masked blocks
      // S^T tiles: hoist all 8 k A-fragments, then run the WMMA chain
      v16h kf[8];
      #pragma unroll
      for (int st = 0; st < 4; ++st) {
        #pragma unroll
        for (int ks = 0; ks < 2; ++ks) {
          const int r  = st * 16 + l16;
          const int db = ks * 32 + hs * 16;
          kf[st * 2 + ks] = cat16(*(const v8h*)&kcur[r * LDA64 + db],
                                  *(const v8h*)&kcur[r * LDA64 + db + 8]);
        }
      }
      v8f St[4];
      #pragma unroll
      for (int st = 0; st < 4; ++st) {
        v8f c = {};
        c = WMMA_F16(kf[st * 2 + 0], qf[0], c);
        c = WMMA_F16(kf[st * 2 + 1], qf[1], c);
        St[st] = c;
      }
      // Causal mask (element (s,t): s > t -> -inf).  Wave-uniform trigger.
      if (s0 + 63 > tw) {
        const int tcol = tw + l16;
        #pragma unroll
        for (int st = 0; st < 4; ++st) {
          const int srow = s0 + st * 16 + hs * 8;
          #pragma unroll
          for (int v2 = 0; v2 < 8; ++v2)
            if (srow + v2 > tcol) St[st][v2] = NEGBIG;
        }
      }
      // Online softmax (exp2 domain; scale folded into q)
      float mx = NEGBIG;
      #pragma unroll
      for (int st = 0; st < 4; ++st)
        #pragma unroll
        for (int v2 = 0; v2 < 8; ++v2) mx = fmaxf(mx, St[st][v2]);
      mx = fmaxf(mx, __shfl_xor(mx, 16, 32));
      const float mnew  = fmaxf(mrow, mx);
      const float alpha = __builtin_amdgcn_exp2f(mrow - mnew);
      float psum = 0.0f;
      #pragma unroll
      for (int st = 0; st < 4; ++st)
        #pragma unroll
        for (int v2 = 0; v2 < 8; ++v2) {
          const float p = __builtin_amdgcn_exp2f(St[st][v2] - mnew);
          St[st][v2] = p;
          psum += p;
        }
      psum += __shfl_xor(psum, 16, 32);
      lrow = lrow * alpha + psum;
      mrow = mnew;
      #pragma unroll
      for (int dt = 0; dt < 4; ++dt)
        #pragma unroll
        for (int v2 = 0; v2 < 8; ++v2) O[dt][v2] *= alpha;

      // Build P^T B-fragments from St (D-layout -> B-layout: one xor-16 exchange)
      v16h pt[2];
      #pragma unroll
      for (int ks = 0; ks < 2; ++ks) {
        v16h p;
        #pragma unroll
        for (int v2 = 0; v2 < 8; ++v2) {
          const float e0  = St[2 * ks][v2];
          const float e1  = St[2 * ks + 1][v2];
          const float e0x = __shfl_xor(e0, 16, 32);
          const float e1x = __shfl_xor(e1, 16, 32);
          p[v2]     = (_Float16)(hs ? e1x : e0);
          p[v2 + 8] = (_Float16)(hs ? e1  : e0x);
        }
        pt[ks] = p;
      }
      // O^T += V^T @ P^T  (hoist all 8 v^T A-fragments, then the WMMA chain)
      v16h vf[8];
      #pragma unroll
      for (int ks = 0; ks < 2; ++ks) {
        #pragma unroll
        for (int dt = 0; dt < 4; ++dt) {
          const int r  = dt * 16 + l16;
          const int sb = ks * 32 + hs * 16;
          vf[ks * 4 + dt] = cat16(*(const v8h*)&vcur[r * LDA64 + sb],
                                  *(const v8h*)&vcur[r * LDA64 + sb + 8]);
        }
      }
      #pragma unroll
      for (int ks = 0; ks < 2; ++ks)
        #pragma unroll
        for (int dt = 0; dt < 4; ++dt)
          O[dt] = WMMA_F16(vf[ks * 4 + dt], pt[ks], O[dt]);
    }
    __syncthreads();
  }

  // Normalize and store f16 head-concat layout [B][T][H*64]; 8 consecutive d's -> b128
  const float inv   = 1.0f / lrow;
  const int   tglob = tw + l16;
  #pragma unroll
  for (int dt = 0; dt < 4; ++dt) {
    v8h pk;
    #pragma unroll
    for (int v2 = 0; v2 < 8; ++v2) pk[v2] = (_Float16)(O[dt][v2] * inv);
    *(v8h*)&aws[(size_t)(b * TSEQ + tglob) * CDIM + h * DHEAD + dt * 16 + hs * 8] = pk;
  }
}

// ---------------------------------------------------------------------------
// Kernel 3: output projection  out = attn_concat @ Wo + bo  (f32 out)
// grid.x = 32 (m-blocks) * 16 (n-blocks of 64).
// A tile is a pure f16 copy -> async double-buffered; B (Wo) needs f32->f16
// conversion -> VGPR-staged pipeline.
// ---------------------------------------------------------------------------
__global__ __launch_bounds__(256) void outproj(
    const _Float16* __restrict__ aws, const float* __restrict__ Wo,
    const float* __restrict__ bo, float* __restrict__ out)
{
  __shared__ _Float16 Ash[KVBUF][128 * LDA32];
  __shared__ _Float16 Bsh[64 * LDA32];

  const int bid = blockIdx.x;
  const int mb  = bid & 31;
  const int nb  = bid >> 5;
  const int m0  = mb * 128, n0 = nb * 64;

  const int tid  = threadIdx.x;
  const int lane = tid & 31;
  const int wv   = __builtin_amdgcn_readfirstlane(tid >> 5);
  const int l16  = lane & 15;
  const int hs   = lane >> 4;

  v8f acc[4] = {};
  float4 bPf[2];

#if USE_ASYNC_LDS
  auto issue_a = [&](int k0, int buf) {
    #pragma unroll
    for (int j = 0; j < 2; ++j) {
      const int idx = tid + j * 256, r = idx >> 2, sg = idx & 3;
      ASYNC_CP_B128(aws + (size_t)(m0 + r) * CDIM + k0 + sg * 8,
                    &Ash[buf][r * LDA32 + sg * 8]);
    }
  };
#else
  v8h aPf[2];
  auto load_a = [&](int k0) {
    #pragma unroll
    for (int j = 0; j < 2; ++j) {
      const int idx = tid + j * 256, r = idx >> 2, sg = idx & 3;
      aPf[j] = *(const v8h*)(aws + (size_t)(m0 + r) * CDIM + k0 + sg * 8);
    }
  };
  auto store_a = [&]() {
    #pragma unroll
    for (int j = 0; j < 2; ++j) {
      const int idx = tid + j * 256, r = idx >> 2, sg = idx & 3;
      *(v8h*)&Ash[0][r * LDA32 + sg * 8] = aPf[j];
    }
  };
#endif
  auto load_b = [&](int k0) {
    #pragma unroll
    for (int j = 0; j < 2; ++j) {
      const int idx = tid + j * 256, kr = idx >> 4, ns = idx & 15;
      bPf[j] = *(const float4*)(Wo + (size_t)(k0 + kr) * CDIM + n0 + ns * 4);
    }
  };
  auto store_b = [&]() {
    #pragma unroll
    for (int j = 0; j < 2; ++j) {
      const int idx = tid + j * 256, kr = idx >> 4, ns = idx & 15;
      Bsh[(ns * 4 + 0) * LDA32 + kr] = (_Float16)bPf[j].x;
      Bsh[(ns * 4 + 1) * LDA32 + kr] = (_Float16)bPf[j].y;
      Bsh[(ns * 4 + 2) * LDA32 + kr] = (_Float16)bPf[j].z;
      Bsh[(ns * 4 + 3) * LDA32 + kr] = (_Float16)bPf[j].w;
    }
  };

  load_b(0);
#if USE_ASYNC_LDS
  issue_a(0, 0);
#else
  load_a(0);
#endif
  for (int k0 = 0; k0 < CDIM; k0 += 32) {
    store_b();
#if USE_ASYNC_LDS
    const int cur = (k0 >> 5) & 1;
    if (k0 + 32 < CDIM) {
      issue_a(k0 + 32, cur ^ 1);
      __builtin_amdgcn_s_wait_asynccnt(2);   // 2 in flight (next); current done
    } else {
      __builtin_amdgcn_s_wait_asynccnt(0);
    }
    __syncthreads();
    if (k0 + 32 < CDIM) load_b(k0 + 32);
#else
    const int cur = 0;
    store_a();
    __syncthreads();
    if (k0 + 32 < CDIM) { load_a(k0 + 32); load_b(k0 + 32); }
#endif

    const int ar = wv * 16 + l16;
    const int ab = hs * 8;
    v16h afrag = cat16(*(const v8h*)&Ash[cur][ar * LDA32 + ab],
                       *(const v8h*)&Ash[cur][ar * LDA32 + ab + 16]);
    v16h bfr[4];
    #pragma unroll
    for (int nt = 0; nt < 4; ++nt) {
      const int bn = nt * 16 + l16;
      const int bb = hs * 8;
      bfr[nt] = cat16(*(const v8h*)&Bsh[bn * LDA32 + bb],
                      *(const v8h*)&Bsh[bn * LDA32 + bb + 16]);
    }
    #pragma unroll
    for (int nt = 0; nt < 4; ++nt)
      acc[nt] = WMMA_F16(afrag, bfr[nt], acc[nt]);
    __syncthreads();
  }

  const int mrow = m0 + wv * 16 + hs * 8;
  #pragma unroll
  for (int nt = 0; nt < 4; ++nt) {
    const int c  = n0 + nt * 16 + l16;
    const float bv = bo[c];
    #pragma unroll
    for (int v2 = 0; v2 < 8; ++v2)
      out[(size_t)(mrow + v2) * CDIM + c] = acc[nt][v2] + bv;
  }
}

// ---------------------------------------------------------------------------
extern "C" void kernel_launch(void* const* d_in, const int* in_sizes, int n_in,
                              void* d_out, int out_size, void* d_ws, size_t ws_size,
                              hipStream_t stream) {
  const float* K  = (const float*)d_in[0];
  const float* Q  = (const float*)d_in[1];
  const float* V  = (const float*)d_in[2];
  const float* Wk = (const float*)d_in[3];
  const float* Wq = (const float*)d_in[4];
  const float* Wv = (const float*)d_in[5];
  const float* Wo = (const float*)d_in[6];
  const float* bo = (const float*)d_in[7];
  float* out = (float*)d_out;

  // Workspace layout (f16): q | k | v^T | attn_out  == 4 * 8 MiB = 32 MiB
  const size_t NQ = (size_t)2 * NH * TSEQ * DHEAD;   // 4,194,304 halves each
  _Float16* qws  = (_Float16*)d_ws;
  _Float16* kws  = qws  + NQ;
  _Float16* vtws = kws  + NQ;
  _Float16* aws  = vtws + NQ;

  proj_qkv<<<3 * NH * 32, 256, 0, stream>>>(K, Q, V, Wk, Wq, Wv, qws, kws, vtws);
  attn    <<<2 * NH * (TSEQ / 128), 256, 0, stream>>>(qws, kws, vtws, aws);
  outproj <<<32 * 16, 256, 0, stream>>>(aws, Wo, bo, out);
}
